// InferenceNetwork_46007689675061
// MI455X (gfx1250) — matrix operand order repaired and checked by reading, not compiled, not run on hardware
//
#include <hip/hip_runtime.h>
#include <hip/hip_bf16.h>

#define TT 32768
#define HH 256
#define G3 768   // 3*H
#define KK 10

typedef __attribute__((ext_vector_type(16))) __bf16 v16bf;
typedef __attribute__((ext_vector_type(8)))  float  v8f;
typedef __attribute__((ext_vector_type(4)))  unsigned int u32x4;
typedef __attribute__((ext_vector_type(8)))  int i32x8;
typedef __attribute__((ext_vector_type(4)))  int i32x4;

union FragAB { v16bf bf; uint4 q[2]; };

__device__ inline unsigned short f2bf(float f) {
    unsigned u = __float_as_uint(f);
    unsigned r = u + 0x7FFFu + ((u >> 16) & 1u);
    return (unsigned short)(r >> 16);
}

__device__ inline v8f wmma_bf16(const FragAB& a, const FragAB& b, v8f c) {
    return __builtin_amdgcn_wmma_f32_16x16x32_bf16(
        false, a.bf, false, b.bf, (short)0, c, false, false);
}

#if defined(__has_builtin)
#if __has_builtin(__builtin_amdgcn_tensor_load_to_lds) && __has_builtin(__builtin_amdgcn_s_wait_tensorcnt)
#define HAVE_TDM 1
#endif
#endif

#ifdef HAVE_TDM
// 2D TDM load: tile_d1 rows x tile_d0 bf16 elements, row-padded in LDS
// (64B data + 16B pad -> 80B LDS row stride, bank-conflict-free ds_load_b128).
__device__ inline void tdm_load_2d(unsigned lds_off, const void* gptr,
                                   unsigned tensor_d0, unsigned tensor_d1,
                                   unsigned tile_d0, unsigned tile_d1,
                                   unsigned d0_stride) {
    unsigned long long ga = (unsigned long long)(size_t)gptr;
    u32x4 g0;
    g0[0] = 1u;                                         // count=1 (valid), no gather
    g0[1] = lds_off;                                    // lds_addr
    g0[2] = (unsigned)(ga & 0xFFFFFFFFu);               // global_addr lo
    g0[3] = (unsigned)((ga >> 32) & 0x01FFFFFFu) | 0x80000000u; // addr hi | type=2
    i32x8 g1;
    // data_size=1(2B) | pad_enable | pad_interval=3(16 dw) | pad_amount=3(4 dw)
    g1[0] = (int)((1u << 16) | (1u << 20) | (3u << 22) | (3u << 25));
    g1[1] = (int)((tensor_d0 & 0xFFFFu) << 16);                         // dim0 lo16
    g1[2] = (int)(((tensor_d0 >> 16) & 0xFFFFu) | ((tensor_d1 & 0xFFFFu) << 16));
    g1[3] = (int)(((tensor_d1 >> 16) & 0xFFFFu) | ((tile_d0 & 0xFFFFu) << 16));
    g1[4] = (int)(tile_d1 & 0xFFFFu);                                   // tile_d1 | tile_d2=0
    g1[5] = (int)d0_stride;                                             // dim0 stride lo32
    g1[6] = 0;
    g1[7] = 0;
    i32x4 z4 = {0, 0, 0, 0};
    i32x8 z8 = {0, 0, 0, 0, 0, 0, 0, 0};
    __builtin_amdgcn_tensor_load_to_lds(g0, g1, z4, z4, z8, 0);
}
#endif

// ---------------- fp32 -> bf16 convert ----------------
__global__ void k_f2bf(const float* __restrict__ s, unsigned short* __restrict__ d, int n) {
    int i = blockIdx.x * blockDim.x + threadIdx.x;
    if (i < n) d[i] = f2bf(s[i]);
}

// ---------------- input projection GEMM ----------------
// out(T,768) = Xbf(T,in) * Wbf(768,in)^T + bias
// 256 threads = 8 waves; block tile M=64, N=64 (wave: 1 m-tile x 2 n-tiles).
// A/B panels (64x32 bf16) staged in LDS by the Tensor Data Mover, double-buffered.
__global__ void __launch_bounds__(256)
k_gemm_proj(const unsigned short* __restrict__ X, const unsigned short* __restrict__ W,
            const float* __restrict__ bias, float* __restrict__ out, int in_dim) {
    __shared__ __align__(16) char shA[2][64 * 80];
    __shared__ __align__(16) char shB[2][64 * 80];

    const int tid = threadIdx.x;
    const int lane = tid & 31, wave = tid >> 5;
    const int lo = lane & 15, hi = lane >> 4;
    const int m0 = blockIdx.x * 64;
    const int n0 = blockIdx.y * 64;
    const int mt = wave & 3;          // m-tile 0..3
    const int ntb = (wave >> 2) * 2;  // n-tile base 0 or 2

    const int nks = in_dim >> 5;
    v8f acc0 = {}, acc1 = {};
    int p = 0;

#ifdef HAVE_TDM
    const unsigned offA0 = (unsigned)(size_t)&shA[0][0];
    const unsigned offA1 = (unsigned)(size_t)&shA[1][0];
    const unsigned offB0 = (unsigned)(size_t)&shB[0][0];
    const unsigned offB1 = (unsigned)(size_t)&shB[1][0];
    if (tid < 32) {
        tdm_load_2d(offA0, X + (size_t)m0 * in_dim, in_dim, TT, 32, 64, in_dim);
        tdm_load_2d(offB0, W + (size_t)n0 * in_dim, in_dim, G3, 32, 64, in_dim);
    }
    for (int ks = 0; ks < nks; ++ks) {
        if (tid < 32) {
            if (ks + 1 < nks) {
                const size_t ko = (size_t)(ks + 1) * 32;
                tdm_load_2d(p ? offA0 : offA1, X + (size_t)m0 * in_dim + ko, in_dim, TT, 32, 64, in_dim);
                tdm_load_2d(p ? offB0 : offB1, W + (size_t)n0 * in_dim + ko, in_dim, G3, 32, 64, in_dim);
                __builtin_amdgcn_s_wait_tensorcnt(2);   // previous stage done, next in flight
            } else {
                __builtin_amdgcn_s_wait_tensorcnt(0);
            }
        }
        __syncthreads();
        {
            FragAB a, b0, b1;
            const char* ap = shA[p] + (mt * 16 + lo) * 80 + hi * 16;
            a.q[0] = *(const uint4*)ap;
            a.q[1] = *(const uint4*)(ap + 32);
            const char* bp0 = shB[p] + ((ntb + 0) * 16 + lo) * 80 + hi * 16;
            b0.q[0] = *(const uint4*)bp0;
            b0.q[1] = *(const uint4*)(bp0 + 32);
            const char* bp1 = shB[p] + ((ntb + 1) * 16 + lo) * 80 + hi * 16;
            b1.q[0] = *(const uint4*)bp1;
            b1.q[1] = *(const uint4*)(bp1 + 32);
            acc0 = wmma_bf16(a, b0, acc0);
            acc1 = wmma_bf16(a, b1, acc1);
        }
        __syncthreads();
        p ^= 1;
    }
#else
    // Fallback: cooperative synchronous staging (256 threads x 16B = one 64x32 panel).
    for (int ks = 0; ks < nks; ++ks) {
        const int row = tid >> 2, ch = tid & 3;
        const size_t kk = (size_t)ks * 32;
        *(uint4*)(shA[0] + row * 80 + ch * 16) =
            *(const uint4*)(X + (size_t)(m0 + row) * in_dim + kk + ch * 8);
        *(uint4*)(shB[0] + row * 80 + ch * 16) =
            *(const uint4*)(W + (size_t)(n0 + row) * in_dim + kk + ch * 8);
        __syncthreads();
        FragAB a, b0, b1;
        const char* ap = shA[0] + (mt * 16 + lo) * 80 + hi * 16;
        a.q[0] = *(const uint4*)ap;
        a.q[1] = *(const uint4*)(ap + 32);
        const char* bp0 = shB[0] + ((ntb + 0) * 16 + lo) * 80 + hi * 16;
        b0.q[0] = *(const uint4*)bp0;
        b0.q[1] = *(const uint4*)(bp0 + 32);
        const char* bp1 = shB[0] + ((ntb + 1) * 16 + lo) * 80 + hi * 16;
        b1.q[0] = *(const uint4*)bp1;
        b1.q[1] = *(const uint4*)(bp1 + 32);
        acc0 = wmma_bf16(a, b0, acc0);
        acc1 = wmma_bf16(a, b1, acc1);
        __syncthreads();
    }
#endif

    // store D tiles: lane holds N=lo, M = v + 8*hi (per 16x16 f32 C/D layout)
    const int mrow = m0 + mt * 16 + hi * 8;
    {
        int n = n0 + (ntb + 0) * 16 + lo;
        float bv = bias[n];
        float* orow = out + (size_t)mrow * G3 + n;
#pragma unroll
        for (int v = 0; v < 8; ++v) orow[(size_t)v * G3] = acc0[v] + bv;
    }
    {
        int n = n0 + (ntb + 1) * 16 + lo;
        float bv = bias[n];
        float* orow = out + (size_t)mrow * G3 + n;
#pragma unroll
        for (int v = 0; v < 8; ++v) orow[(size_t)v * G3] = acc1[v] + bv;
    }
}

// ---------------- sequential GRU scan, one block per direction ----------------
// 768 threads = 24 waves; each wave owns 2 row-tiles (32 rows) of w_hh as
// register-resident bf16 WMMA A-fragments. h is broadcast via LDS as B operand.
__global__ void __launch_bounds__(768)
k_gru_scan(const unsigned short* __restrict__ whh_f, const unsigned short* __restrict__ whh_b,
           const float* __restrict__ ib_f, const float* __restrict__ ib_b,
           const float* __restrict__ bn_f, const float* __restrict__ bn_b,
           float* __restrict__ out) {
    const int dir = blockIdx.x;
    const unsigned short* whh = dir ? whh_b : whh_f;
    const float* ibuf = dir ? ib_b : ib_f;
    const float* bn   = dir ? bn_b : bn_f;

    __shared__ __align__(16) unsigned short sh_h[HH]; // h in bf16 (B operand source)
    __shared__ float sh_wv[G3];                       // matvec results

    int tid = threadIdx.x;
    int lane = tid & 31, wave = tid >> 5;
    int lo = lane & 15, hi = lane >> 4;

    // Preload w_hh fragments into registers: 2 tiles x 8 K-steps x 16 bf16.
    FragAB wfr[2][8];
#pragma unroll
    for (int tt = 0; tt < 2; ++tt) {
        int row = (wave * 2 + tt) * 16 + lo;
        const unsigned short* wr = whh + (size_t)row * HH;
#pragma unroll
        for (int s = 0; s < 8; ++s) {
            int ka = s * 32 + hi * 8;
            wfr[tt][s].q[0] = *(const uint4*)(wr + ka);
            wfr[tt][s].q[1] = *(const uint4*)(wr + ka + 16);
        }
    }

    if (tid < HH) sh_h[tid] = 0;
    float hprev = 0.f, bnv = 0.f, ir = 0.f, iz = 0.f, ig = 0.f;
    int t = dir ? (TT - 1) : 0;
    if (tid < HH) {
        bnv = bn[tid];
        const float* ip = ibuf + (size_t)t * G3 + tid;
        ir = ip[0]; iz = ip[HH]; ig = ip[2 * HH];
    }
    __syncthreads();

    const int tileA = wave * 2, tileB = wave * 2 + 1;
    for (int step = 0; step < TT; ++step) {
        // ---- matvec phase: wv = w_hh @ h via WMMA (column 0 of D is the result) ----
        v8f acc0 = {}, acc1 = {};
        const uint4* hq = (const uint4*)sh_h;
#pragma unroll
        for (int s = 0; s < 8; ++s) {
            FragAB b;
            b.q[0] = hq[s * 4 + hi];           // K = s*32 + 8*hi .. +7
            b.q[1] = hq[s * 4 + hi + 2];       // K = s*32 + 16 + 8*hi .. +7
            acc0 = wmma_bf16(wfr[0][s], b, acc0);
            acc1 = wmma_bf16(wfr[1][s], b, acc1);
        }
        if (lo == 0) {                         // lanes 0 and 16 hold column N=0
            int mb = hi * 8;
#pragma unroll
            for (int v = 0; v < 8; ++v) {
                sh_wv[tileA * 16 + mb + v] = acc0[v];
                sh_wv[tileB * 16 + mb + v] = acc1[v];
            }
        }
        __syncthreads();

        // ---- gate phase ----
        if (tid < HH) {
            float hr = sh_wv[tid], hz = sh_wv[HH + tid], hg = sh_wv[2 * HH + tid];
            float r = 1.f / (1.f + __expf(-(ir + hr)));
            float z = 1.f / (1.f + __expf(-(iz + hz)));
            float ga = ig + r * (hg + bnv);
            float e2 = __expf(2.f * ga);
            float g = (e2 - 1.f) / (e2 + 1.f);
            float hn = (1.f - z) * g + z * hprev;
            hprev = hn;
            sh_h[tid] = f2bf(hn);
            out[(size_t)t * (2 * HH) + dir * HH + tid] = hn;
            int tn = dir ? t - 1 : t + 1;
            if (step + 1 < TT) {               // prefetch next step's i_all
                const float* ip = ibuf + (size_t)tn * G3 + tid;
                ir = ip[0]; iz = ip[HH]; ig = ip[2 * HH];
            }
            t = tn;
        }
        __syncthreads();
    }
}

// ---------------- output head ----------------
__global__ void k_head(const float* __restrict__ h, const float* __restrict__ mw,
                       const float* __restrict__ mb, const float* __restrict__ sw,
                       const float* __restrict__ sb, float* __restrict__ out) {
    int idx = blockIdx.x * blockDim.x + threadIdx.x;
    if (idx >= TT * KK) return;
    int t = idx / KK, k = idx % KK;
    const float4* hr = (const float4*)(h + (size_t)t * 512);
    const float4* m4 = (const float4*)(mw + (size_t)k * 512);
    const float4* s4 = (const float4*)(sw + (size_t)k * 512);
    float am = 0.f, as = 0.f;
    for (int i = 0; i < 128; ++i) {
        float4 hv = hr[i], mv = m4[i], sv = s4[i];
        am += hv.x * mv.x + hv.y * mv.y + hv.z * mv.z + hv.w * mv.w;
        as += hv.x * sv.x + hv.y * sv.y + hv.z * sv.z + hv.w * sv.w;
    }
    am += mb[k];
    as += sb[k];
    out[idx] = am;
    out[TT * KK + idx] = fmaxf(as, 0.f) + __logf(1.f + __expf(-fabsf(as)));
}

// ---------------- workspace layout (bytes) ----------------
static constexpr size_t OFF_XBF  = 0;                                  // T*512 bf16
static constexpr size_t OFF_WIHF = OFF_XBF  + (size_t)TT * 512 * 2;    // 768*512 bf16
static constexpr size_t OFF_WIHB = OFF_WIHF + (size_t)G3 * 512 * 2;
static constexpr size_t OFF_WHHF = OFF_WIHB + (size_t)G3 * 512 * 2;    // 768*256 bf16
static constexpr size_t OFF_WHHB = OFF_WHHF + (size_t)G3 * HH * 2;
static constexpr size_t OFF_IBF  = OFF_WHHB + (size_t)G3 * HH * 2;     // T*768 f32
static constexpr size_t OFF_IBB  = OFF_IBF  + (size_t)TT * G3 * 4;
static constexpr size_t OFF_OUT0 = OFF_IBB  + (size_t)TT * G3 * 4;     // T*512 f32
static constexpr size_t OFF_OUT1 = OFF_OUT0 + (size_t)TT * 512 * 4;

extern "C" void kernel_launch(void* const* d_in, const int* in_sizes, int n_in,
                              void* d_out, int out_size, void* d_ws, size_t ws_size,
                              hipStream_t stream) {
    (void)in_sizes; (void)n_in; (void)out_size; (void)ws_size;

    const float* x = (const float*)d_in[0];
    char* ws = (char*)d_ws;
    unsigned short* xbf  = (unsigned short*)(ws + OFF_XBF);
    unsigned short* wihf = (unsigned short*)(ws + OFF_WIHF);
    unsigned short* wihb = (unsigned short*)(ws + OFF_WIHB);
    unsigned short* whhf = (unsigned short*)(ws + OFF_WHHF);
    unsigned short* whhb = (unsigned short*)(ws + OFF_WHHB);
    float* ibf  = (float*)(ws + OFF_IBF);
    float* ibb  = (float*)(ws + OFF_IBB);
    float* out0 = (float*)(ws + OFF_OUT0);
    float* out1 = (float*)(ws + OFF_OUT1);

    for (int l = 0; l < 2; ++l) {
        const int in_dim = l ? 512 : 64;
        const float* X = l ? out0 : x;
        const int base = 1 + l * 8;
        const float* w_ih_f = (const float*)d_in[base + 0];
        const float* w_hh_f = (const float*)d_in[base + 1];
        const float* b_f    = (const float*)d_in[base + 2];
        const float* bn_f   = (const float*)d_in[base + 3];
        const float* w_ih_b = (const float*)d_in[base + 4];
        const float* w_hh_b = (const float*)d_in[base + 5];
        const float* b_b    = (const float*)d_in[base + 6];
        const float* bn_b   = (const float*)d_in[base + 7];

        const int nX = TT * in_dim;
        k_f2bf<<<(nX + 255) / 256, 256, 0, stream>>>(X, xbf, nX);
        const int nW = G3 * in_dim;
        k_f2bf<<<(nW + 255) / 256, 256, 0, stream>>>(w_ih_f, wihf, nW);
        k_f2bf<<<(nW + 255) / 256, 256, 0, stream>>>(w_ih_b, wihb, nW);
        const int nH = G3 * HH;
        k_f2bf<<<(nH + 255) / 256, 256, 0, stream>>>(w_hh_f, whhf, nH);
        k_f2bf<<<(nH + 255) / 256, 256, 0, stream>>>(w_hh_b, whhb, nH);

        dim3 gg(TT / 64, G3 / 64);
        k_gemm_proj<<<gg, 256, 0, stream>>>(xbf, wihf, b_f, ibf, in_dim);
        k_gemm_proj<<<gg, 256, 0, stream>>>(xbf, wihb, b_b, ibb, in_dim);

        float* outL = l ? out1 : out0;
        k_gru_scan<<<2, 768, 0, stream>>>(whhf, whhb, ibf, ibb, bn_f, bn_b, outL);
    }

    const float* mu_w = (const float*)d_in[17];
    const float* mu_b = (const float*)d_in[18];
    const float* sg_w = (const float*)d_in[19];
    const float* sg_b = (const float*)d_in[20];
    k_head<<<(TT * KK + 255) / 256, 256, 0, stream>>>(out1, mu_w, mu_b, sg_w, sg_b, (float*)d_out);
}
